// PanopPseudoLabels_18786186952785
// MI455X (gfx1250) — compile-verified, matrix operand order checked by reading, not run on hardware
//
#include <hip/hip_runtime.h>
#include <hip/hip_bf16.h>

typedef __attribute__((ext_vector_type(16))) _Float16 v16h;
typedef __attribute__((ext_vector_type(8)))  float    v8f;
typedef __attribute__((ext_vector_type(2)))  float    v2f;
typedef __attribute__((ext_vector_type(4)))  _Float16 h4;

#define HAVE_WMMA_F32X4 __has_builtin(__builtin_amdgcn_wmma_f32_16x16x4_f32)

#define BB      2
#define HH      512
#define WW      512
#define HWSZ    (HH * WW)            // 262144
#define NPIX    (BB * HWSZ)          // 524288
#define TOPK    200
#define KPAD    208                  // 13 tiles of 16
#define NTILE   13
#define RAD_    25
#define WIN     51
#define THR_    0.1f
#define MIN_PXL_ 64
#define CAP     65536

// ---------------- workspace byte offsets ----------------
#define WS_CANDV   ((size_t)0)                       // float[CAP]
#define WS_CANDI   ((size_t)(CAP * 4))               // int[CAP]
#define WS_CNT     ((size_t)(CAP * 8))               // int[1]
#define WS_BIDS    (WS_CNT   + 64)                   // int[TOPK]
#define WS_CYS     (WS_BIDS  + 1024)
#define WS_CXS     (WS_CYS   + 1024)
#define WS_VALID   (WS_CXS   + 1024)
#define WS_SEL     (WS_VALID + 1024)
#define WS_LABEL   (WS_SEL   + 1024)                 // int[KPAD]
#define WS_ANYC    (WS_LABEL + 1024)                 // int[BB]
#define WS_AMAT    (WS_ANYC  + 64)                   // f32: v2f[BB*NTILE*32] (6656B) / f16: h4[BB*KPAD]
#define WS_STATS   (WS_AMAT  + 8192)                 // uint[3*BB*(TOPK+1)]
#define WS_ROWMAX  (WS_STATS + 8192)                 // float[NPIX]
#define WS_INS     (WS_ROWMAX + (size_t)NPIX * 4)    // int[NPIX]

// ---------------- init: zero center output, stats, counter ----------------
__global__ void k_init(float* out_center, unsigned int* stats, int* counter) {
    int i = blockIdx.x * blockDim.x + threadIdx.x;
    if (i < NPIX) out_center[i] = 0.0f;
    if (i < 3 * BB * (TOPK + 1)) stats[i] = 0u;
    if (i == 0) *counter = 0;
}

// ---------------- NMS pass 1: threshold + 7-wide row max ----------------
__global__ void k_rowmax(const float* __restrict__ hmp, float* __restrict__ rowmax) {
    int i = blockIdx.x * blockDim.x + threadIdx.x;
    if (i >= NPIX) return;
    int x = i % WW;
    int rowbase = i - x;
    float m = -1e30f;
    #pragma unroll
    for (int d = -3; d <= 3; ++d) {
        int xx = x + d;
        if (xx >= 0 && xx < WW) {
            float v = hmp[rowbase + xx];
            v = (v > THR_) ? v : -1.0f;
            m = fmaxf(m, v);
        }
    }
    rowmax[i] = m;
}

// ---------------- NMS pass 2: 7-wide col max, peak test, compact candidates ----------------
__global__ void k_peaks(const float* __restrict__ hmp, const float* __restrict__ rowmax,
                        float* cand_val, int* cand_idx, int* counter) {
    int i = blockIdx.x * blockDim.x + threadIdx.x;
    if (i >= NPIX) return;
    int b = i / HWSZ;
    int pin = i % HWSZ;
    int y = pin / WW, x = pin % WW;
    float t = hmp[i];
    t = (t > THR_) ? t : -1.0f;
    float m = -1e30f;
    #pragma unroll
    for (int d = -3; d <= 3; ++d) {
        int yy = y + d;
        if (yy >= 0 && yy < HH) m = fmaxf(m, rowmax[b * HWSZ + yy * WW + x]);
    }
    if (t == m && t >= THR_) {
        int pos = atomicAdd(counter, 1);
        if (pos < CAP) { cand_val[pos] = t; cand_idx[pos] = i; }
    }
}

// ---------------- top-K selection (single block, deterministic) ----------------
__global__ void k_topk(float* cand_val, const int* __restrict__ cand_idx, const int* counter,
                       int* bids, int* cys, int* cxs, int* valid) {
    __shared__ float sv[256];
    __shared__ int   si[256];
    __shared__ int   sp[256];
    int tid = threadIdx.x;
    int n = *counter;
    if (n > CAP) n = CAP;
    for (int r = 0; r < TOPK; ++r) {
        float bv = -1e30f; int bi = 0x7fffffff; int bp = -1;
        for (int i = tid; i < n; i += 256) {
            float v = cand_val[i];
            int   id = cand_idx[i];
            if (v > bv || (v == bv && id < bi)) { bv = v; bi = id; bp = i; }
        }
        sv[tid] = bv; si[tid] = bi; sp[tid] = bp;
        __syncthreads();
        for (int s = 128; s > 0; s >>= 1) {
            if (tid < s) {
                if (sv[tid + s] > sv[tid] || (sv[tid + s] == sv[tid] && si[tid + s] < si[tid])) {
                    sv[tid] = sv[tid + s]; si[tid] = si[tid + s]; sp[tid] = sp[tid + s];
                }
            }
            __syncthreads();
        }
        if (tid == 0) {
            if (sp[0] >= 0 && sv[0] >= THR_) {
                int id = si[0];
                bids[r] = id / HWSZ;
                cys[r]  = (id % HWSZ) / WW;
                cxs[r]  = id % WW;
                valid[r] = 1;
                cand_val[sp[0]] = -1e30f;   // remove winner
            } else {
                bids[r] = 0; cys[r] = 0; cxs[r] = 0; valid[r] = 0;
            }
        }
        __syncthreads();
    }
}

// ---------------- per-center filter: mixmask + 51x51 thing-pixel count ----------------
__global__ void k_filter(const int* __restrict__ semantic, const float* __restrict__ mix,
                         const int* bids, const int* cys, const int* cxs,
                         const int* valid, int* selected) {
    __shared__ int ssum[256];
    __shared__ int sflag;
    int k = blockIdx.x;
    int tid = threadIdx.x;
    if (tid == 0) {
        int f = 0;
        if (valid[k]) {
            float mv = mix[bids[k] * HWSZ + cys[k] * WW + cxs[k]];
            if (mv < 0.5f) f = 1;
        }
        sflag = f;
    }
    __syncthreads();
    int cnt = 0;
    if (sflag) {
        int b = bids[k], cy = cys[k], cx = cxs[k];
        int y0 = max(0, cy - RAD_), y1 = min(HH, cy + RAD_ + 1);
        int x0 = max(0, cx - RAD_), x1 = min(WW, cx + RAD_ + 1);
        int wx = x1 - x0;
        int wN = (y1 - y0) * wx;
        for (int i = tid; i < wN; i += 256) {
            int yy = y0 + i / wx;
            int xx = x0 + i % wx;
            int s = semantic[b * HWSZ + yy * WW + xx];
            cnt += (s >= 11 && s <= 18) ? 1 : 0;
        }
    }
    ssum[tid] = cnt;
    __syncthreads();
    for (int s = 128; s > 0; s >>= 1) {
        if (tid < s) ssum[tid] += ssum[tid + s];
        __syncthreads();
    }
    if (tid == 0) selected[k] = (sflag && ssum[0] >= MIN_PXL_) ? 1 : 0;
}

// ---------------- labels, any-center flags, pre-expanded WMMA A-matrix ----------------
__global__ void k_prep(const int* bids, const int* cys, const int* cxs, const int* selected,
                       int* label, int* anyc, void* Abuf) {
    int tid = threadIdx.x;
    if (tid == 0) {
        int per[BB];
        for (int b = 0; b < BB; ++b) { per[b] = 0; anyc[b] = 0; }
        for (int k = 0; k < TOPK; ++k) {
            if (selected[k]) {
                int b = bids[k];
                per[b] += 1;
                label[k] = per[b];      // local index within batch + 1
                anyc[b] = 1;
            } else label[k] = 0;
        }
        for (int k = TOPK; k < KPAD; ++k) label[k] = 0;
    }
    __syncthreads();
#if HAVE_WMMA_F32X4
    // Full per-lane A layout for V_WMMA_F32_16X16X4_F32:
    // lanes 0-15 (M=lane):   V0 = K0 = -2*cy, V1 = K1 = -2*cx
    // lanes 16-31 (M=lane-16): V0 = K2 = ||c||^2 (or BIG), V1 = K3 = 0
    v2f* Afull = (v2f*)Abuf;
    for (int t = tid; t < BB * NTILE * 32; t += blockDim.x) {
        int b    = t / (NTILE * 32);
        int r    = t % (NTILE * 32);
        int tile = r / 32;
        int lane = r % 32;
        int kk   = tile * 16 + (lane & 15);
        bool v = (kk < TOPK) && selected[kk] && (bids[kk] == b);
        v2f a;
        if (lane < 16) {
            a[0] = v ? (float)(-2 * cys[kk]) : 0.0f;
            a[1] = v ? (float)(-2 * cxs[kk]) : 0.0f;
        } else {
            a[0] = v ? (float)(cys[kk] * cys[kk] + cxs[kk] * cxs[kk]) : 1.0e10f;
            a[1] = 0.0f;
        }
        Afull[t] = a;
    }
#else
    // Compact f16 rows [-2cy, -2cx, hi, lo] with hi*256+lo = ||c||^2 (f16-exact split)
    _Float16* Acmp = (_Float16*)Abuf;
    for (int t = tid; t < BB * KPAD; t += blockDim.x) {
        int b = t / KPAD;
        int kk = t % KPAD;
        _Float16* a = Acmp + (size_t)t * 4;
        bool v = (kk < TOPK) && selected[kk] && (bids[kk] == b);
        if (v) {
            int cy = cys[kk], cx = cxs[kk];
            int c2 = cy * cy + cx * cx;
            a[0] = (_Float16)(float)(-2 * cy);
            a[1] = (_Float16)(float)(-2 * cx);
            a[2] = (_Float16)(float)(c2 >> 8);
            a[3] = (_Float16)(float)(c2 & 255);
        } else {
            a[0] = (_Float16)0.0f;
            a[1] = (_Float16)0.0f;
            a[2] = (_Float16)65504.0f;
            a[3] = (_Float16)0.0f;
        }
    }
#endif
}

// ---------------- Gaussian max-splat (bit-pattern atomicMax, values >= 0) ----------------
__global__ void k_splat(const int* bids, const int* cys, const int* cxs, const int* selected,
                        float* center) {
    int k = blockIdx.x;
    if (!selected[k]) return;
    int b = bids[k], cy = cys[k], cx = cxs[k];
    const float inv2s2 = 1.0f / 128.0f;          // 1/(2*8*8)
    for (int i = threadIdx.x; i < WIN * WIN; i += blockDim.x) {
        int dy = i / WIN - RAD_;
        int dx = i % WIN - RAD_;
        int y = cy + dy, x = cx + dx;
        if (y >= 0 && y < HH && x >= 0 && x < WW) {
            float v = expf(-(float)(dy * dy + dx * dx) * inv2s2);
            atomicMax((unsigned int*)&center[b * HWSZ + y * WW + x], __float_as_uint(v));
        }
    }
}

// ---------------- WMMA nearest-center grouping: 16 pixels x 208 centers per wave ----------------
__global__ void __launch_bounds__(256) k_group(const float* __restrict__ offsets,
                                               const int* __restrict__ semantic,
                                               const void* __restrict__ Abuf,
                                               const int* __restrict__ label,
                                               const int* __restrict__ anyc,
                                               int* __restrict__ ins) {
    int wave = (blockIdx.x * blockDim.x + threadIdx.x) >> 5;   // wave32
    int lane = threadIdx.x & 31;
    int n  = lane & 15;        // pixel column within tile
    int up = lane >> 4;        // 0: M=0..7, 1: M=8..15 of D
    int base = wave * 16;      // first pixel of this group (NPIX % 16 == 0)
    int b = base / HWSZ;
    int pin = (base + n) % HWSZ;
    int y = pin / WW, x = pin % WW;

    float offy = offsets[(size_t)b * 2 * HWSZ + pin];
    float offx = offsets[(size_t)b * 2 * HWSZ + HWSZ + pin];
    float py = (float)y + offy;
    float px = (float)x + offx;

    float best = 3.0e38f;
    int bestk = 0;

#if HAVE_WMMA_F32X4
    // B column n = [py, px, 1, 0]^T : lanes 0-15 -> (K0,K1), lanes 16-31 -> (K2,K3)
    v2f bm;
    bm[0] = (up == 0) ? py : 1.0f;
    bm[1] = (up == 0) ? px : 0.0f;

    const v2f* A = (const v2f*)Abuf + (size_t)b * (NTILE * 32);
    v2f ar[NTILE];
    #pragma unroll
    for (int t = 0; t < NTILE; ++t) ar[t] = A[t * 32 + lane];   // pipelined, unconditional

    #pragma unroll
    for (int t = 0; t < NTILE; ++t) {
        v8f c = {};
        // score[M=center, N=pixel] = ||c||^2 - 2 c.p  (argmin-equivalent to squared distance)
        v8f d = __builtin_amdgcn_wmma_f32_16x16x4_f32(
            false, ar[t], false, bm, (short)0, c, false, false);
        #pragma unroll
        for (int v = 0; v < 8; ++v) {
            float s = d[v];
            int k = t * 16 + up * 8 + v;
            if (s < best) { best = s; bestk = k; }   // ascending k -> first-min tie-break
        }
    }
    const float validThr = 1.0e9f;
#else
    v16h bm = {};
    if (up == 0) {
        bm[0] = (_Float16)py;
        bm[1] = (_Float16)px;
        bm[2] = (_Float16)256.0f;
        bm[3] = (_Float16)1.0f;
    }
    const h4* A = (const h4*)Abuf + (size_t)b * KPAD;
    h4 ar[NTILE];
    #pragma unroll
    for (int t = 0; t < NTILE; ++t) ar[t] = A[t * 16 + n];      // hoisted loads

    #pragma unroll
    for (int t = 0; t < NTILE; ++t) {
        v16h am = {};
        if (up == 0) { am[0] = ar[t][0]; am[1] = ar[t][1]; am[2] = ar[t][2]; am[3] = ar[t][3]; }
        v8f c = {};
        v8f d = __builtin_amdgcn_wmma_f32_16x16x32_f16(
            false, am, false, bm, (short)0, c, false, false);
        #pragma unroll
        for (int v = 0; v < 8; ++v) {
            float s = d[v];
            int k = t * 16 + up * 8 + v;
            if (s < best) { best = s; bestk = k; }
        }
    }
    const float validThr = 8.0e6f;
#endif

    // combine the two half-lanes that hold the same pixel column
    float obest = __shfl_xor(best, 16, 32);
    int   okey  = __shfl_xor(bestk, 16, 32);
    if (obest < best || (obest == best && okey < bestk)) { best = obest; bestk = okey; }

    if (up == 0) {
        int s = semantic[b * HWSZ + pin];
        bool thing = (s >= 11) && (s <= 18);
        int out = 0;
        if (thing && anyc[b] && best < validThr) out = label[bestk];
        ins[b * HWSZ + pin] = out;
    }
}

// ---------------- per-instance integer stats (deterministic) ----------------
__global__ void k_stats(const int* __restrict__ ins, unsigned int* __restrict__ stats) {
    __shared__ unsigned int scnt[TOPK + 1], ssy[TOPK + 1], ssx[TOPK + 1];
    int tid = threadIdx.x;
    for (int i = tid; i <= TOPK; i += blockDim.x) { scnt[i] = 0u; ssy[i] = 0u; ssx[i] = 0u; }
    __syncthreads();
    int i = blockIdx.x * blockDim.x + tid;
    int b = i / HWSZ;                       // uniform per block (HWSZ % 256 == 0)
    int pin = i % HWSZ;
    int y = pin / WW, x = pin % WW;
    int s = ins[i];
    if (s > 0) {
        atomicAdd(&scnt[s], 1u);
        atomicAdd(&ssy[s], (unsigned int)y);
        atomicAdd(&ssx[s], (unsigned int)x);
    }
    __syncthreads();
    for (int k = tid; k <= TOPK; k += blockDim.x) {
        if (scnt[k] > 0u) {
            unsigned int* g = stats + ((size_t)b * (TOPK + 1) + k) * 3;
            atomicAdd(&g[0], scnt[k]);
            atomicAdd(&g[1], ssy[k]);
            atomicAdd(&g[2], ssx[k]);
        }
    }
}

// ---------------- final outputs: offsets, weights ----------------
__global__ void k_final(const int* __restrict__ ins, const unsigned int* __restrict__ stats,
                        float* __restrict__ out) {
    int i = blockIdx.x * blockDim.x + threadIdx.x;
    if (i >= NPIX) return;
    int b = i / HWSZ;
    int pin = i % HWSZ;
    int y = pin / WW, x = pin % WW;
    int s = ins[i];
    float oy = 0.0f, ox = 0.0f, w = 0.0f;
    if (s > 0) {
        const unsigned int* g = stats + ((size_t)b * (TOPK + 1) + s) * 3;
        float c = fmaxf((float)g[0], 1.0f);
        oy = (float)g[1] / c - (float)y;
        ox = (float)g[2] / c - (float)x;
        w = 1.0f;
    }
    float* off = out + NPIX;                         // [B,2,H,W]
    off[(size_t)b * 2 * HWSZ + pin] = oy;
    off[(size_t)b * 2 * HWSZ + HWSZ + pin] = ox;
    float* cw = out + NPIX + (size_t)BB * 2 * HWSZ;  // center_weights [B,H,W]
    cw[i] = 1.0f;
    float* ow = cw + NPIX;                           // offset_weights [B,H,W]
    ow[i] = w;
}

extern "C" void kernel_launch(void* const* d_in, const int* in_sizes, int n_in,
                              void* d_out, int out_size, void* d_ws, size_t ws_size,
                              hipStream_t stream) {
    (void)in_sizes; (void)n_in; (void)out_size; (void)ws_size;
    const float* ctr_hmp  = (const float*)d_in[0];   // [B,1,H,W]
    const float* offsets  = (const float*)d_in[1];   // [B,2,H,W]
    const int*   semantic = (const int*)d_in[2];     // [B,H,W]
    const float* mixmasks = (const float*)d_in[3];   // [B,1,H,W]
    float* out = (float*)d_out;

    char* w = (char*)d_ws;
    float*        cand_val = (float*)(w + WS_CANDV);
    int*          cand_idx = (int*)(w + WS_CANDI);
    int*          counter  = (int*)(w + WS_CNT);
    int*          bids     = (int*)(w + WS_BIDS);
    int*          cys      = (int*)(w + WS_CYS);
    int*          cxs      = (int*)(w + WS_CXS);
    int*          valid    = (int*)(w + WS_VALID);
    int*          selected = (int*)(w + WS_SEL);
    int*          label    = (int*)(w + WS_LABEL);
    int*          anyc     = (int*)(w + WS_ANYC);
    void*         Abuf     = (void*)(w + WS_AMAT);
    unsigned int* stats    = (unsigned int*)(w + WS_STATS);
    float*        rowmax   = (float*)(w + WS_ROWMAX);
    int*          ins      = (int*)(w + WS_INS);

    dim3 blk(256);
    dim3 gpix((NPIX + 255) / 256);

    k_init  <<<gpix, blk, 0, stream>>>(out, stats, counter);
    k_rowmax<<<gpix, blk, 0, stream>>>(ctr_hmp, rowmax);
    k_peaks <<<gpix, blk, 0, stream>>>(ctr_hmp, rowmax, cand_val, cand_idx, counter);
    k_topk  <<<1,    blk, 0, stream>>>(cand_val, cand_idx, counter, bids, cys, cxs, valid);
    k_filter<<<TOPK, blk, 0, stream>>>(semantic, mixmasks, bids, cys, cxs, valid, selected);
    k_prep  <<<1,    blk, 0, stream>>>(bids, cys, cxs, selected, label, anyc, Abuf);
    k_splat <<<TOPK, blk, 0, stream>>>(bids, cys, cxs, selected, out);
    // WMMA grouping: NPIX/16 waves = 32768 waves = 4096 blocks of 8 waves
    k_group <<<NPIX / 16 / 8, blk, 0, stream>>>(offsets, semantic, Abuf, label, anyc, ins);
    k_stats <<<gpix, blk, 0, stream>>>(ins, stats);
    k_final <<<gpix, blk, 0, stream>>>(ins, stats, out);
}
